// VQVAE_6605659701615
// MI455X (gfx1250) — compile-verified, matrix operand order checked by reading, not run on hardware
//
#include <hip/hip_runtime.h>
#include <hip/hip_bf16.h>

typedef __attribute__((ext_vector_type(16))) __bf16 v16bf;
typedef __attribute__((ext_vector_type(8)))  float  v8f;

// ---------------- problem sizes ----------------
constexpr int kB     = 32;
constexpr int kNpix1 = kB * 128 * 128;   // 524288  (h1 / g1 pixels)
constexpr int kNpix2 = kB * 64 * 64;     // 131072  (h2 / z pixels == N rows for VQ)
constexpr int kNpix0 = kB * 256 * 256;   // 2097152 (g2 / recon pixels)

// ---------------- workspace layout (float units) ----------------
constexpr size_t OFF_H1     = 0;                                  // [B,128,128,64] f32
constexpr size_t OFF_H2     = OFF_H1 + (size_t)kNpix1 * 64;       // [B,64,64,128]  f32
constexpr size_t OFF_Z      = OFF_H2 + (size_t)kNpix2 * 128;      // [N,64] f32
constexpr size_t OFF_G1     = OFF_Z  + (size_t)kNpix2 * 64;       // [B,128,128,64] f32
constexpr size_t OFF_G2     = OFF_G1 + (size_t)kNpix1 * 64;       // [B,256,256,32] f32
constexpr size_t OFF_STATS  = OFF_G2 + (size_t)kNpix0 * 32;       // 1024
constexpr size_t OFF_SCALES = OFF_STATS + 1024;                   // 1024
constexpr size_t OFF_VQACC  = OFF_SCALES + 1024;                  // 1 (+7 pad)
constexpr size_t OFF_CNORM  = OFF_VQACC + 8;                      // 512
// bf16 regions (float units = bf16 elements / 2); all offsets multiple of 8 -> 32B aligned
constexpr size_t OFF_ZBF    = OFF_CNORM + 512;                    // [N,64] bf16
constexpr size_t OFF_QBF    = OFF_ZBF + (size_t)kNpix2 * 32;      // [N,64] bf16
constexpr size_t OFF_W2BF   = OFF_QBF + (size_t)kNpix2 * 32;      // swizzled [18][8][32][16] bf16
constexpr size_t OFF_W3BF   = OFF_W2BF + 36864;                   // swizzled [4][4][32][16]  bf16
constexpr size_t OFF_CBT    = OFF_W3BF + 4096;                    // swizzled [2][32][32][16] bf16
constexpr size_t OFF_WD1    = OFF_CBT + 16384;                    // swizzled [4][8][4][32][16] bf16
constexpr size_t OFF_WD2    = OFF_WD1 + 32768;                    // swizzled [4][8][2][32][16] bf16

// stats / scale sub-offsets
constexpr size_t S_SUM1 = OFF_STATS + 0,   S_SSQ1 = OFF_STATS + 128;
constexpr size_t S_SUM2 = OFF_STATS + 256, S_SSQ2 = OFF_STATS + 384;
constexpr size_t S_SUMD1 = OFF_STATS + 512, S_SSQD1 = OFF_STATS + 640;
constexpr size_t S_SUMD2 = OFF_STATS + 768, S_SSQD2 = OFF_STATS + 896;
constexpr size_t C_SC1 = OFF_SCALES + 0,   C_SH1 = OFF_SCALES + 128;
constexpr size_t C_SC2 = OFF_SCALES + 256, C_SH2 = OFF_SCALES + 384;
constexpr size_t C_SCD1 = OFF_SCALES + 512, C_SHD1 = OFF_SCALES + 640;
constexpr size_t C_SCD2 = OFF_SCALES + 768, C_SHD2 = OFF_SCALES + 896;

// d_out: [recon 2097152][vq_loss 1][idx 131072]
constexpr size_t OUT_VQ  = (size_t)kNpix0;
constexpr size_t OUT_IDX = (size_t)kNpix0 + 1;

// ---------------- device helpers ----------------
__device__ __forceinline__ float lrelu(float v) { return v > 0.f ? v : 0.2f * v; }

__device__ __forceinline__ v8f zero8() {
  v8f z = {0.f, 0.f, 0.f, 0.f, 0.f, 0.f, 0.f, 0.f};
  return z;
}

// A fragment: 16x32 bf16 tile, element (m,k) at A[m*lda + k] (two 16B runs/lane -> ds_load_b128).
__device__ __forceinline__ v16bf load_a_frag(const __bf16* A, int lda) {
  const int lane = threadIdx.x & 31;
  const int m  = lane & 15;
  const int kh = (lane >> 4) << 3;
  v16bf a;
#pragma unroll
  for (int h = 0; h < 16; ++h) {
    const int k = ((h < 8) ? h : (h + 8)) + kh;
    a[h] = A[m * lda + k];
  }
  return a;
}

// B fragment from pre-swizzled storage: one 32B vector load per lane (coalesced).
__device__ __forceinline__ v16bf load_b_sw(const __bf16* tile) {
  const int lane = threadIdx.x & 31;
  return *(const v16bf*)(tile + lane * 16);
}

// inverse fragment mapping for a 32x16 B tile: element (kk,n) -> (lane, h)
__device__ __forceinline__ int b_sw_index(int kk, int n) {
  const int lane = n + (((kk >> 3) & 1) << 4);
  const int h    = (kk & 7) + (((kk >> 4) & 1) << 3);
  return lane * 16 + h;
}

__device__ __forceinline__ v8f wmma_bf16(v16bf a, v16bf b, v8f c) {
  return __builtin_amdgcn_wmma_f32_16x16x32_bf16(false, a, false, b, (short)0, c, false, false);
}

// ---------------- small utility kernels ----------------
__global__ void k_zero_stats(float* stats, float* vqacc) {
  const int t = blockIdx.x * blockDim.x + threadIdx.x;
  if (t < 1024) stats[t] = 0.f;
  if (t == 0) vqacc[0] = 0.f;
}

// per-channel sum/sumsq over NHWC buffer (consecutive threads = consecutive channels)
__global__ void k_stats(const float* __restrict__ buf, int npix, int C,
                        float* __restrict__ sum, float* __restrict__ ssq) {
  const int tid = threadIdx.x;
  const int c = tid % C;
  const int slot = tid / C;
  const int slots = blockDim.x / C;
  float s = 0.f, q = 0.f;
  for (int p = blockIdx.x * slots + slot; p < npix; p += gridDim.x * slots) {
    const float v = buf[(size_t)p * C + c];
    s += v; q += v * v;
  }
  __shared__ float ls[128], lq[128];
  if (tid < C) { ls[tid] = 0.f; lq[tid] = 0.f; }
  __syncthreads();
  atomicAdd(&ls[c], s);
  atomicAdd(&lq[c], q);
  __syncthreads();
  if (tid < C) { atomicAdd(&sum[tid], ls[tid]); atomicAdd(&ssq[tid], lq[tid]); }
}

__global__ void k_bn_finalize(const float* __restrict__ sum, const float* __restrict__ ssq,
                              const float* __restrict__ g, const float* __restrict__ b,
                              float count, int C,
                              float* __restrict__ scale, float* __restrict__ shift) {
  const int t = threadIdx.x;
  if (t < C) {
    const float m = sum[t] / count;
    const float v = ssq[t] / count - m * m;
    const float rs = rsqrtf(v + 1e-5f);
    scale[t] = g[t] * rs;
    shift[t] = b[t] - m * g[t] * rs;
  }
}

// ---------------- weight prep (fp32 -> bf16 swizzled fragment layouts) ----------------
// conv2 weights -> [ks(18)][ntile(8)][32][16]
__global__ void k_prep_w2(const float* __restrict__ w, __bf16* __restrict__ dst) {
  const int e = blockIdx.x * blockDim.x + threadIdx.x;   // (kglob, o)
  if (e >= 576 * 128) return;
  const int kg = e >> 7, o = e & 127;
  const int tap = kg >> 6, c = kg & 63;
  const float v = w[((size_t)o * 64 + c) * 9 + tap];
  const int ks = kg >> 5, kk = kg & 31;
  const int nt = o >> 4, n = o & 15;
  dst[(size_t)(ks * 8 + nt) * 512 + b_sw_index(kk, n)] = (__bf16)v;
}

// conv3 weights -> [ks(4)][ntile(4)][32][16]
__global__ void k_prep_w3(const float* __restrict__ w, __bf16* __restrict__ dst) {
  const int e = blockIdx.x * blockDim.x + threadIdx.x;   // (k, o)
  if (e >= 128 * 64) return;
  const int kg = e >> 6, o = e & 63;
  const float v = w[(size_t)o * 128 + kg];
  const int ks = kg >> 5, kk = kg & 31;
  const int nt = o >> 4, n = o & 15;
  dst[(size_t)(ks * 4 + nt) * 512 + b_sw_index(kk, n)] = (__bf16)v;
}

// codebook^T -> [kt(2)][ct(32)][32][16] ; plus ||e||^2
__global__ void k_prep_cb(const float* __restrict__ cb, __bf16* __restrict__ cbT,
                          float* __restrict__ cnorm) {
  const int e = blockIdx.x * blockDim.x + threadIdx.x;
  if (e < 64 * 512) {
    const int d = e >> 9, code = e & 511;
    const float v = cb[(size_t)code * 64 + d];
    const int kt = d >> 5, kk = d & 31;
    const int ct = code >> 4, n = code & 15;
    cbT[(size_t)(kt * 32 + ct) * 512 + b_sw_index(kk, n)] = (__bf16)v;
  }
  if (e < 512) {
    float s = 0.f;
    for (int d = 0; d < 64; ++d) { const float v = cb[(size_t)e * 64 + d]; s += v * v; }
    cnorm[e] = s;
  }
}

// transposed-conv weights, per output parity -> [p][ks(8)][ntile][32][16]
__global__ void k_prep_wdec(const float* __restrict__ w, __bf16* __restrict__ dst, int Cout) {
  const int e = blockIdx.x * blockDim.x + threadIdx.x;
  const int perP = 256 * Cout;
  if (e >= 4 * perP) return;
  const int p = e / perP;
  const int r = e - p * perP;
  const int kg = r / Cout;
  const int o = r - kg * Cout;
  const int tslot = kg >> 6, c = kg & 63;
  const int ty = tslot >> 1, tx = tslot & 1;
  const int py = p >> 1, px = p & 1;
  const int ky = (py == 0) ? (ty == 0 ? 1 : -1) : (ty == 0 ? 0 : 2);
  const int kx = (px == 0) ? (tx == 0 ? 1 : -1) : (tx == 0 ? 0 : 2);
  float v = 0.f;
  if (ky >= 0 && kx >= 0) v = w[(((size_t)o * 64 + c) * 3 + ky) * 3 + kx];
  const int ntiles = Cout >> 4;
  const int ks = kg >> 5, kk = kg & 31;
  const int nt = o >> 4, n = o & 15;
  dst[(size_t)((p * 8 + ks) * ntiles + nt) * 512 + b_sw_index(kk, n)] = (__bf16)v;
}

// ---------------- encoder ----------------
// conv1: Cin=1, 3x3 s2 -> h1 NHWC [B,128,128,64] (K=9: VALU, memory bound)
__global__ void k_conv1(const float* __restrict__ x, const float* __restrict__ w1,
                        const float* __restrict__ b1, float* __restrict__ h1) {
  __shared__ float w1s[576];
  __shared__ float b1s[64];
  for (int i = threadIdx.x; i < 576; i += blockDim.x) w1s[i] = w1[i];
  if (threadIdx.x < 64) b1s[threadIdx.x] = b1[threadIdx.x];
  __syncthreads();
  const int pix = blockIdx.x * blockDim.x + threadIdx.x;
  const int b = pix >> 14;
  const int oy = (pix >> 7) & 127;
  const int ox = pix & 127;
  float t[9];
#pragma unroll
  for (int ky = 0; ky < 3; ++ky)
#pragma unroll
    for (int kx = 0; kx < 3; ++kx) {
      const int iy = 2 * oy + ky - 1, ix = 2 * ox + kx - 1;
      float v = 0.f;
      if (iy >= 0 && iy < 256 && ix >= 0 && ix < 256)
        v = x[((size_t)b * 256 + iy) * 256 + ix];
      t[ky * 3 + kx] = v;
    }
  float* out = h1 + (size_t)pix * 64;
#pragma unroll 4
  for (int c = 0; c < 64; ++c) {
    float acc = b1s[c];
#pragma unroll
    for (int j = 0; j < 9; ++j) acc += t[j] * w1s[c * 9 + j];
    out[c] = acc;
  }
}

// conv2: 64->128, 3x3 s2, implicit GEMM K=576, M-tile=32 (2 accumulators/wave).
// block = 256 (8 waves = 8 N-tiles of 16 ch), 32 pixels per block.
__global__ void k_conv2_wmma(const float* __restrict__ h1, const float* __restrict__ sc,
                             const float* __restrict__ sh, const __bf16* __restrict__ w2bf,
                             const float* __restrict__ b2, float* __restrict__ h2) {
  __shared__ __bf16 As[32 * 608];
  const int tid = threadIdx.x;
  const int m0 = blockIdx.x * 32;
  for (int row = 0; row < 32; ++row) {
    const int pix = m0 + row;
    const int b = pix >> 12;
    const int oy = (pix >> 6) & 63;
    const int ox = pix & 63;
    for (int k = tid; k < 576; k += 256) {
      const int tap = k >> 6, c = k & 63;
      const int ky = tap / 3, kx = tap - ky * 3;
      const int iy = 2 * oy + ky - 1, ix = 2 * ox + kx - 1;
      float v = 0.f;
      if (iy >= 0 && iy < 128 && ix >= 0 && ix < 128)
        v = lrelu(h1[(((size_t)b * 128 + iy) * 128 + ix) * 64 + c] * sc[c] + sh[c]);
      As[row * 608 + k] = (__bf16)v;
    }
  }
  __syncthreads();
  const int wave = tid >> 5, lane = tid & 31;
  const int n = wave * 16 + (lane & 15);
  v8f acc0 = zero8(), acc1 = zero8();
  for (int ks = 0; ks < 18; ++ks) {
    if (ks + 1 < 18) __builtin_prefetch(w2bf + (size_t)((ks + 1) * 8 + wave) * 512, 0, 1);
    const v16bf b = load_b_sw(w2bf + (size_t)(ks * 8 + wave) * 512);
    acc0 = wmma_bf16(load_a_frag(&As[ks * 32], 608), b, acc0);
    acc1 = wmma_bf16(load_a_frag(&As[16 * 608 + ks * 32], 608), b, acc1);
  }
  const float bias = b2[n];
#pragma unroll
  for (int r = 0; r < 8; ++r) {
    const int mrow = r + ((lane >> 4) << 3);
    h2[(size_t)(m0 + mrow) * 128 + n] = acc0[r] + bias;
    h2[(size_t)(m0 + 16 + mrow) * 128 + n] = acc1[r] + bias;
  }
}

// conv3: 1x1, 128->64 GEMM, M-tile=32. BN2+lrelu fused. Writes z (f32) and zbf (bf16).
// block = 128 (4 waves), 32 pixels per block.
__global__ void k_conv3_wmma(const float* __restrict__ h2, const float* __restrict__ sc,
                             const float* __restrict__ sh, const __bf16* __restrict__ w3bf,
                             const float* __restrict__ b3,
                             float* __restrict__ z, __bf16* __restrict__ zbf) {
  __shared__ __bf16 As[32 * 128];
  const int tid = threadIdx.x;   // 128
  const int m0 = blockIdx.x * 32;
  for (int row = 0; row < 32; ++row) {
    const float v = lrelu(h2[(size_t)(m0 + row) * 128 + tid] * sc[tid] + sh[tid]);
    As[row * 128 + tid] = (__bf16)v;
  }
  __syncthreads();
  const int wave = tid >> 5, lane = tid & 31;
  const int n = wave * 16 + (lane & 15);
  v8f acc0 = zero8(), acc1 = zero8();
  for (int ks = 0; ks < 4; ++ks) {
    const v16bf b = load_b_sw(w3bf + (size_t)(ks * 4 + wave) * 512);
    acc0 = wmma_bf16(load_a_frag(&As[ks * 32], 128), b, acc0);
    acc1 = wmma_bf16(load_a_frag(&As[16 * 128 + ks * 32], 128), b, acc1);
  }
  const float bias = b3[n];
#pragma unroll
  for (int r = 0; r < 8; ++r) {
    const int mrow = r + ((lane >> 4) << 3);
    const float v0 = acc0[r] + bias;
    const float v1 = acc1[r] + bias;
    z[(size_t)(m0 + mrow) * 64 + n] = v0;
    zbf[(size_t)(m0 + mrow) * 64 + n] = (__bf16)v0;
    z[(size_t)(m0 + 16 + mrow) * 64 + n] = v1;
    zbf[(size_t)(m0 + 16 + mrow) * 64 + n] = (__bf16)v1;
  }
}

// ---------------- vector quantization ----------------
// argmin_e(||e||^2 - 2 z.e) via WMMA; running min in the striped C layout, then
// cross-lane shfl_xor reduce within the 16-lane halves of the wave32.
__global__ void k_vq_wmma(const __bf16* __restrict__ zbf, const float* __restrict__ z,
                          const __bf16* __restrict__ cbT, const float* __restrict__ cnorm,
                          const float* __restrict__ cb, __bf16* __restrict__ qbf,
                          float* __restrict__ idx_out, float* __restrict__ vqacc) {
  const int tid = threadIdx.x;
  const int wave = tid >> 5, lane = tid & 31;
  const int m0 = (blockIdx.x * (blockDim.x >> 5) + wave) * 16;
  const v16bf a0 = load_a_frag(zbf + (size_t)m0 * 64, 64);
  const v16bf a1 = load_a_frag(zbf + (size_t)m0 * 64 + 32, 64);
  float bd[8]; int bi[8];
#pragma unroll
  for (int r = 0; r < 8; ++r) { bd[r] = 3.4e38f; bi[r] = 0; }
  for (int ct = 0; ct < 32; ++ct) {                 // 32 tiles of 16 codes
    if (ct + 1 < 32) __builtin_prefetch(cbT + (size_t)(ct + 1) * 512, 0, 1);
    v8f acc = zero8();
    acc = wmma_bf16(a0, load_b_sw(cbT + (size_t)ct * 512), acc);
    acc = wmma_bf16(a1, load_b_sw(cbT + (size_t)(32 + ct) * 512), acc);
    const int code = ct * 16 + (lane & 15);
    const float cn = cnorm[code];
#pragma unroll
    for (int r = 0; r < 8; ++r) {
      const float d = cn - 2.f * acc[r];
      if (d < bd[r]) { bd[r] = d; bi[r] = code; }   // codes strictly increasing per lane
    }
  }
#pragma unroll
  for (int r = 0; r < 8; ++r) {
    for (int msk = 1; msk < 16; msk <<= 1) {
      const float od = __shfl_xor(bd[r], msk, 32);
      const int   oi = __shfl_xor(bi[r], msk, 32);
      if (od < bd[r] || (od == bd[r] && oi < bi[r])) { bd[r] = od; bi[r] = oi; }
    }
  }
  const int half = lane >> 4, lr = lane & 15;
  if (lr < 8) {   // 16 active lanes handle the 16 rows
    const int r = lr;
    const int row = m0 + half * 8 + r;
    const int idx = bi[r];
    idx_out[row] = (float)idx;
    float s = 0.f;
    for (int d = 0; d < 64; ++d) {
      const float e  = cb[(size_t)idx * 64 + d];
      const float zv = z[(size_t)row * 64 + d];
      qbf[(size_t)row * 64 + d] = (__bf16)e;        // straight-through forward value
      const float df = e - zv;
      s += df * df;
    }
    atomicAdd(vqacc, s);
  }
}

__global__ void k_vq_finalize(const float* __restrict__ vqacc, float* __restrict__ out) {
  if (threadIdx.x == 0 && blockIdx.x == 0)
    out[0] = vqacc[0] * (1.25f / ((float)kNpix2 * 64.f));   // (1 + beta) * mean
}

// ---------------- decoder (transposed convs: parity-decomposed implicit GEMM, K=256) ----------------
// decT1: 64->64, input qbf bf16 [B,64,64,64], output g1 f32 [B,128,128,64]. block=128, M-tile=32.
__global__ void k_dect1_wmma(const __bf16* __restrict__ qbf, const __bf16* __restrict__ wPar,
                             const float* __restrict__ bias, float* __restrict__ g1) {
  __shared__ __bf16 As[32 * 272];
  const int tid = threadIdx.x;   // 128
  const int m0 = blockIdx.x * 32;
  const int p = m0 >> 17;        // 131072 rows per parity
  const int py = p >> 1, px = p & 1;
  for (int row = 0; row < 32; ++row) {
    const int rem = (m0 + row) & 131071;
    const int b = rem >> 12;
    const int ys = (rem >> 6) & 63;
    const int xs = rem & 63;
    for (int k = tid; k < 256; k += 128) {
      const int tslot = k >> 6, c = k & 63;
      const int ty = tslot >> 1, tx = tslot & 1;
      int iy = ys, ix = xs;
      bool ok = true;
      if (py == 0) { if (ty != 0) ok = false; } else { iy = ys + ty; if (iy > 63) ok = false; }
      if (px == 0) { if (tx != 0) ok = false; } else { ix = xs + tx; if (ix > 63) ok = false; }
      __bf16 v = (__bf16)0.f;
      if (ok) v = qbf[(((size_t)b * 64 + iy) * 64 + ix) * 64 + c];
      As[row * 272 + k] = v;
    }
  }
  __syncthreads();
  const int wave = tid >> 5, lane = tid & 31;
  const int n = wave * 16 + (lane & 15);
  v8f acc0 = zero8(), acc1 = zero8();
  for (int ks = 0; ks < 8; ++ks) {
    const v16bf b = load_b_sw(wPar + (size_t)((p * 8 + ks) * 4 + wave) * 512);
    acc0 = wmma_bf16(load_a_frag(&As[ks * 32], 272), b, acc0);
    acc1 = wmma_bf16(load_a_frag(&As[16 * 272 + ks * 32], 272), b, acc1);
  }
  const float bs = bias[n];
#pragma unroll
  for (int r = 0; r < 8; ++r) {
    const int mrow = r + ((lane >> 4) << 3);
#pragma unroll
    for (int half = 0; half < 2; ++half) {
      const int rem = (m0 + half * 16 + mrow) & 131071;
      const int b = rem >> 12;
      const int ys = (rem >> 6) & 63;
      const int xs = rem & 63;
      const int oy = 2 * ys + py, ox = 2 * xs + px;
      const float v = (half ? acc1[r] : acc0[r]) + bs;
      g1[(((size_t)b * 128 + oy) * 128 + ox) * 64 + n] = v;
    }
  }
}

// decT2: 64->32, input g1 f32 (dbn1+lrelu fused), output g2 f32 [B,256,256,32]. block=64, M-tile=32.
__global__ void k_dect2_wmma(const float* __restrict__ g1, const float* __restrict__ sc,
                             const float* __restrict__ sh, const __bf16* __restrict__ wPar,
                             const float* __restrict__ bias, float* __restrict__ g2) {
  __shared__ __bf16 As[32 * 272];
  const int tid = threadIdx.x;   // 64
  const int m0 = blockIdx.x * 32;
  const int p = m0 >> 19;        // 524288 rows per parity
  const int py = p >> 1, px = p & 1;
  for (int row = 0; row < 32; ++row) {
    const int rem = (m0 + row) & 524287;
    const int b = rem >> 14;
    const int ys = (rem >> 7) & 127;
    const int xs = rem & 127;
    for (int k = tid; k < 256; k += 64) {
      const int tslot = k >> 6, c = k & 63;
      const int ty = tslot >> 1, tx = tslot & 1;
      int iy = ys, ix = xs;
      bool ok = true;
      if (py == 0) { if (ty != 0) ok = false; } else { iy = ys + ty; if (iy > 127) ok = false; }
      if (px == 0) { if (tx != 0) ok = false; } else { ix = xs + tx; if (ix > 127) ok = false; }
      float v = 0.f;
      if (ok) v = lrelu(g1[(((size_t)b * 128 + iy) * 128 + ix) * 64 + c] * sc[c] + sh[c]);
      As[row * 272 + k] = (__bf16)v;
    }
  }
  __syncthreads();
  const int wave = tid >> 5, lane = tid & 31;
  const int n = wave * 16 + (lane & 15);   // Cout = 32
  v8f acc0 = zero8(), acc1 = zero8();
  for (int ks = 0; ks < 8; ++ks) {
    const v16bf b = load_b_sw(wPar + (size_t)((p * 8 + ks) * 2 + wave) * 512);
    acc0 = wmma_bf16(load_a_frag(&As[ks * 32], 272), b, acc0);
    acc1 = wmma_bf16(load_a_frag(&As[16 * 272 + ks * 32], 272), b, acc1);
  }
  const float bs = bias[n];
#pragma unroll
  for (int r = 0; r < 8; ++r) {
    const int mrow = r + ((lane >> 4) << 3);
#pragma unroll
    for (int half = 0; half < 2; ++half) {
      const int rem = (m0 + half * 16 + mrow) & 524287;
      const int b = rem >> 14;
      const int ys = (rem >> 7) & 127;
      const int xs = rem & 127;
      const int oy = 2 * ys + py, ox = 2 * xs + px;
      const float v = (half ? acc1[r] : acc0[r]) + bs;
      g2[(((size_t)b * 256 + oy) * 256 + ox) * 32 + n] = v;
    }
  }
}

// dec3: 1x1 32->1 + sigmoid, dbn2+lrelu fused. Pure streaming stage.
__global__ void k_dec3(const float* __restrict__ g2, const float* __restrict__ sc,
                       const float* __restrict__ sh, const float* __restrict__ w,
                       const float* __restrict__ b, float* __restrict__ recon) {
  __shared__ float ws_[32], scs[32], shs[32];
  if (threadIdx.x < 32) {
    ws_[threadIdx.x] = w[threadIdx.x];
    scs[threadIdx.x] = sc[threadIdx.x];
    shs[threadIdx.x] = sh[threadIdx.x];
  }
  __syncthreads();
  const int pix = blockIdx.x * blockDim.x + threadIdx.x;
  const float* in = g2 + (size_t)pix * 32;
  float acc = b[0];
#pragma unroll
  for (int j = 0; j < 32; ++j) acc += lrelu(in[j] * scs[j] + shs[j]) * ws_[j];
  recon[pix] = 1.f / (1.f + __expf(-acc));
}

// ---------------- host launch ----------------
extern "C" void kernel_launch(void* const* d_in, const int* in_sizes, int n_in,
                              void* d_out, int out_size, void* d_ws, size_t ws_size,
                              hipStream_t stream) {
  (void)in_sizes; (void)n_in; (void)out_size; (void)ws_size;
  const float* x        = (const float*)d_in[0];
  const float* enc_w1   = (const float*)d_in[1];
  const float* enc_b1   = (const float*)d_in[2];
  const float* bn1_g    = (const float*)d_in[3];
  const float* bn1_b    = (const float*)d_in[4];
  const float* enc_w2   = (const float*)d_in[5];
  const float* enc_b2   = (const float*)d_in[6];
  const float* bn2_g    = (const float*)d_in[7];
  const float* bn2_b    = (const float*)d_in[8];
  const float* enc_w3   = (const float*)d_in[9];
  const float* enc_b3   = (const float*)d_in[10];
  const float* codebook = (const float*)d_in[11];
  const float* dec_w1   = (const float*)d_in[12];
  const float* dec_b1   = (const float*)d_in[13];
  const float* dbn1_g   = (const float*)d_in[14];
  const float* dbn1_b   = (const float*)d_in[15];
  const float* dec_w2   = (const float*)d_in[16];
  const float* dec_b2   = (const float*)d_in[17];
  const float* dbn2_g   = (const float*)d_in[18];
  const float* dbn2_b   = (const float*)d_in[19];
  const float* dec_w3   = (const float*)d_in[20];
  const float* dec_b3   = (const float*)d_in[21];

  float* out = (float*)d_out;
  float* wsf = (float*)d_ws;

  float* h1 = wsf + OFF_H1;
  float* h2 = wsf + OFF_H2;
  float* z  = wsf + OFF_Z;
  float* g1 = wsf + OFF_G1;
  float* g2 = wsf + OFF_G2;
  float* stats = wsf + OFF_STATS;
  float* vqacc = wsf + OFF_VQACC;
  float* cnorm = wsf + OFF_CNORM;
  __bf16* zbf  = (__bf16*)(wsf + OFF_ZBF);
  __bf16* qbf  = (__bf16*)(wsf + OFF_QBF);
  __bf16* w2bf = (__bf16*)(wsf + OFF_W2BF);
  __bf16* w3bf = (__bf16*)(wsf + OFF_W3BF);
  __bf16* cbT  = (__bf16*)(wsf + OFF_CBT);
  __bf16* wd1  = (__bf16*)(wsf + OFF_WD1);
  __bf16* wd2  = (__bf16*)(wsf + OFF_WD2);

  // init + weight prep
  k_zero_stats<<<4, 256, 0, stream>>>(stats, vqacc);
  k_prep_w2<<<288, 256, 0, stream>>>(enc_w2, w2bf);
  k_prep_w3<<<32, 256, 0, stream>>>(enc_w3, w3bf);
  k_prep_cb<<<128, 256, 0, stream>>>(codebook, cbT, cnorm);
  k_prep_wdec<<<256, 256, 0, stream>>>(dec_w1, wd1, 64);
  k_prep_wdec<<<128, 256, 0, stream>>>(dec_w2, wd2, 32);

  // encoder
  k_conv1<<<kNpix1 / 256, 256, 0, stream>>>(x, enc_w1, enc_b1, h1);
  k_stats<<<2048, 256, 0, stream>>>(h1, kNpix1, 64, wsf + S_SUM1, wsf + S_SSQ1);
  k_bn_finalize<<<1, 128, 0, stream>>>(wsf + S_SUM1, wsf + S_SSQ1, bn1_g, bn1_b,
                                       (float)kNpix1, 64, wsf + C_SC1, wsf + C_SH1);
  k_conv2_wmma<<<kNpix2 / 32, 256, 0, stream>>>(h1, wsf + C_SC1, wsf + C_SH1, w2bf, enc_b2, h2);
  k_stats<<<2048, 256, 0, stream>>>(h2, kNpix2, 128, wsf + S_SUM2, wsf + S_SSQ2);
  k_bn_finalize<<<1, 128, 0, stream>>>(wsf + S_SUM2, wsf + S_SSQ2, bn2_g, bn2_b,
                                       (float)kNpix2, 128, wsf + C_SC2, wsf + C_SH2);
  k_conv3_wmma<<<kNpix2 / 32, 128, 0, stream>>>(h2, wsf + C_SC2, wsf + C_SH2, w3bf, enc_b3, z, zbf);

  // quantize
  k_vq_wmma<<<kNpix2 / (16 * 8), 256, 0, stream>>>(zbf, z, cbT, cnorm, codebook, qbf,
                                                   out + OUT_IDX, vqacc);
  k_vq_finalize<<<1, 32, 0, stream>>>(vqacc, out + OUT_VQ);

  // decoder
  k_dect1_wmma<<<kNpix1 / 32, 128, 0, stream>>>(qbf, wd1, dec_b1, g1);
  k_stats<<<2048, 256, 0, stream>>>(g1, kNpix1, 64, wsf + S_SUMD1, wsf + S_SSQD1);
  k_bn_finalize<<<1, 128, 0, stream>>>(wsf + S_SUMD1, wsf + S_SSQD1, dbn1_g, dbn1_b,
                                       (float)kNpix1, 64, wsf + C_SCD1, wsf + C_SHD1);
  k_dect2_wmma<<<kNpix0 / 32, 64, 0, stream>>>(g1, wsf + C_SCD1, wsf + C_SHD1, wd2, dec_b2, g2);
  k_stats<<<2048, 256, 0, stream>>>(g2, kNpix0, 32, wsf + S_SUMD2, wsf + S_SSQD2);
  k_bn_finalize<<<1, 128, 0, stream>>>(wsf + S_SUMD2, wsf + S_SSQD2, dbn2_g, dbn2_b,
                                       (float)kNpix0, 32, wsf + C_SCD2, wsf + C_SHD2);
  k_dec3<<<kNpix0 / 256, 256, 0, stream>>>(g2, wsf + C_SCD2, wsf + C_SHD2, dec_w3, dec_b3, out);
}
